// TimeSeriesInParformer_86947317940430
// MI455X (gfx1250) — matrix operand order repaired and checked by reading, not compiled, not run on hardware
//
#include <hip/hip_runtime.h>
#include <hip/hip_bf16.h>
#include <cstdint>
#include <cstddef>

// ---------------------------------------------------------------------------
// MI455X (gfx1250) TimeSeriesInParformer.
// - all GEMMs on v_wmma_f32_16x16x32_bf16 (bf16 in, fp32 accumulate)
// - LDS tiles filled with global_load_async_to_lds_b128 (ASYNCcnt),
//   double-buffered: async copy of tile t+1 overlaps WMMA of tile t
// - GEMM block tile 128x128 (wave tile 32x64 -> 8 WMMA / k-step)
// - flash attention: V tile transposed once per tile in LDS so all WMMA
//   fragment reads are contiguous ds_load_b128
// ---------------------------------------------------------------------------

typedef __attribute__((ext_vector_type(16))) __bf16 v16bf;
typedef __attribute__((ext_vector_type(8)))  float  v8f;

__device__ __forceinline__ __bf16 f2bf(float f) {
  unsigned int u = __builtin_bit_cast(unsigned int, f);
  unsigned int r = u + 0x7fffu + ((u >> 16) & 1u);   // round-to-nearest-even
  unsigned short h = (unsigned short)(r >> 16);
  return __builtin_bit_cast(__bf16, h);
}
__device__ __forceinline__ __bf16 bf_zero() {
  return __builtin_bit_cast(__bf16, (unsigned short)0);
}

// K index inside a 16-bit A-fragment (ISA 7.12.2, 16-bit A 16x32):
// lanes 0-15: VGPR0-3 -> K=0..7 (pairs), VGPR4-7 -> K=16..23 ; lanes16-31: +8
__device__ __forceinline__ int a_frag_k(int j, int hl) {
  int p = j >> 1;
  int base = (p < 4) ? (2 * p) : (16 + 2 * (p - 4));
  return base + 8 * hl + (j & 1);
}

__device__ __forceinline__ unsigned lds_off(const void* p) {
  return (unsigned)(size_t)p;      // LDS aperture: offset lives in addr[31:0]
}
__device__ __forceinline__ void async_b128(unsigned lds, unsigned goff,
                                           const void* base) {
  asm volatile("global_load_async_to_lds_b128 %0, %1, %2"
               :: "v"(lds), "v"(goff), "s"(base) : "memory");
}
__device__ __forceinline__ void wait_async0() {
  asm volatile("s_wait_asynccnt 0x0" ::: "memory");
}

// ---------------------------------------------------------------------------
// GEMM: C[M,N] = A[M,K] (bf16, row major, lda) * B[N,K]^T (bf16, row major,
// ldb) + bias[N].  K, lda, ldb multiples of 32 (padded where needed).
// flags: bit0 = accumulate into fp32 C, bit1 = ReLU.
// C (fp32) and Cbf (bf16 mirror) each optional (nullptr to skip).
// Block: 256 threads = 8 waves; tile 128x128x32; wave tile 32x64 (acc[2][4]);
// double-buffered async LDS staging.
// ---------------------------------------------------------------------------
#define GBM 128
#define GBN 128
#define GBK 32

__global__ __launch_bounds__(256)
void gemm_bf16_wmma(const __bf16* __restrict__ A, const __bf16* __restrict__ Bm,
                    const float* __restrict__ bias,
                    float* __restrict__ C, __bf16* __restrict__ Cbf,
                    int M, int Nn, int K, int lda, int ldb, int ldc, int flags) {
  __shared__ __bf16 As[2][GBM][GBK + 8];   // row stride 80B (16B aligned)
  __shared__ __bf16 Bs[2][GBN][GBK + 8];
  const int tid  = threadIdx.x;
  const int wave = tid >> 5;
  const int lane = tid & 31;
  const int hl   = lane >> 4;
  const int l15  = lane & 15;
  const int wm   = wave & 3;               // 4 waves x 32 rows
  const int wn   = wave >> 2;              // 2 waves x 64 cols
  const int m0   = blockIdx.x * GBM;
  const int n0   = blockIdx.y * GBN;

  auto issue = [&](int k0, int buf) {
    // A tile: 128 rows x 64B = 512 x 16B chunks ; B tile: 512 chunks
    #pragma unroll
    for (int c = tid; c < 512; c += 256) {
      int r = c >> 2, sub = c & 3;
      unsigned goff = (unsigned)(((size_t)(m0 + r) * lda + k0 + sub * 8) * 2);
      async_b128(lds_off(&As[buf][r][sub * 8]), goff, A);
    }
    #pragma unroll
    for (int c = tid; c < 512; c += 256) {
      int r = c >> 2, sub = c & 3;
      unsigned goff = (unsigned)(((size_t)(n0 + r) * ldb + k0 + sub * 8) * 2);
      async_b128(lds_off(&Bs[buf][r][sub * 8]), goff, Bm);
    }
  };

  v8f acc[2][4] = {};
  const int T = K / GBK;
  issue(0, 0);
  for (int t = 0; t < T; ++t) {
    const int cur = t & 1;
    wait_async0();
    __syncthreads();
    if (t + 1 < T) issue((t + 1) * GBK, cur ^ 1);

    v16bf af[2];
    #pragma unroll
    for (int fm = 0; fm < 2; ++fm) {
      int m = wm * 32 + fm * 16 + l15;
      #pragma unroll
      for (int j = 0; j < 16; ++j) af[fm][j] = As[cur][m][hl * 16 + j];
    }
    #pragma unroll
    for (int fn = 0; fn < 4; ++fn) {
      v16bf bfr;
      int n = wn * 64 + fn * 16 + l15;
      int kb = hl * 16;
      #pragma unroll
      for (int j = 0; j < 16; ++j) bfr[j] = Bs[cur][n][kb + j];
      #pragma unroll
      for (int fm = 0; fm < 2; ++fm)
        acc[fm][fn] = __builtin_amdgcn_wmma_f32_16x16x32_bf16(
            false, af[fm], false, bfr, (short)0, acc[fm][fn], false, false);
    }
    __syncthreads();
  }

  // epilogue: C/D layout: lane l15 = col, vgpr r -> row r + 8*hl
  #pragma unroll
  for (int fm = 0; fm < 2; ++fm) {
    #pragma unroll
    for (int fn = 0; fn < 4; ++fn) {
      int n = n0 + wn * 64 + fn * 16 + l15;
      float bv = bias ? bias[n] : 0.f;
      #pragma unroll
      for (int r = 0; r < 8; ++r) {
        int m = m0 + wm * 32 + fm * 16 + r + 8 * hl;
        float v = acc[fm][fn][r] + bv;
        if (flags & 2) v = fmaxf(v, 0.f);
        size_t idx = (size_t)m * ldc + n;
        if (C)  { if (flags & 1) C[idx] += v; else C[idx] = v; }
        if (Cbf) Cbf[idx] = f2bf(v);
      }
    }
  }
}

// ---------------------------------------------------------------------------
// Fused attention (flash style). qkv bf16 (token n = s*B + b, row = 3E).
// Block = 8 waves, each wave 16 query rows (128-row strip); 32 keys/tile,
// K/V tiles double-buffered via async LDS copies; V transposed in LDS so
// P@V fragment reads are contiguous.  Grid: (S/128, B, H).  Output bf16.
// ---------------------------------------------------------------------------
__global__ __launch_bounds__(256)
void attn_flash_wmma(const __bf16* __restrict__ qkv, __bf16* __restrict__ out,
                     int Bsz, int S, int E, int H) {
  const int dh = 64;
  __shared__ __bf16 Kt[2][32][72];       // row stride 144B (16B aligned)
  __shared__ __bf16 Vt[2][32][72];
  __shared__ __bf16 VtT[64][40];         // V tile transposed (d x key)
  __shared__ __bf16 Ps[8][16][40];       // per-wave P relayout scratch

  const int tid  = threadIdx.x;
  const int wave = tid >> 5;
  const int lane = tid & 31;
  const int hl   = lane >> 4;
  const int l15  = lane & 15;
  const int b    = blockIdx.y;
  const int h    = blockIdx.z;
  const int q0   = blockIdx.x * 128 + wave * 16;
  const size_t ld = (size_t)3 * E;

  const __bf16* qp = qkv + (size_t)h * dh;
  const __bf16* kp = qkv + (size_t)E + (size_t)h * dh;
  const __bf16* vp = qkv + (size_t)2 * E + (size_t)h * dh;

  const int rr  = tid >> 3;              // 32 rows
  const int sub = tid & 7;               // 8 x 16B chunks per 64-col row
  auto issueKV = [&](int kt, int buf) {
    unsigned goff = (unsigned)((((size_t)(kt + rr) * Bsz + b) * ld + sub * 8) * 2);
    async_b128(lds_off(&Kt[buf][rr][sub * 8]), goff, kp);
    async_b128(lds_off(&Vt[buf][rr][sub * 8]), goff, vp);
  };

  // Q fragment: 16 rows x 64 (two 16x32 A-frags), kept in registers
  v16bf qf[2];
  #pragma unroll
  for (int c = 0; c < 2; ++c) {
    size_t row = ((size_t)(q0 + l15) * Bsz + b) * ld;
    #pragma unroll
    for (int j = 0; j < 16; ++j)
      qf[c][j] = qp[row + c * 32 + hl * 16 + j];
  }

  v8f o[4] = {};
  float mrow[8], lrow[8];
  #pragma unroll
  for (int r = 0; r < 8; ++r) { mrow[r] = -1e30f; lrow[r] = 0.f; }
  const float scale = 0.125f;            // 1/sqrt(64)

  const int NT = S / 32;
  issueKV(0, 0);
  for (int t = 0; t < NT; ++t) {
    const int cur = t & 1;
    wait_async0();
    __syncthreads();                     // K/V tile [cur] ready
    if (t + 1 < NT) issueKV((t + 1) * 32, cur ^ 1);

    // transpose V tile (contiguous b128 read, scattered b16 writes)
    {
      int key = tid >> 3, d8 = (tid & 7) * 8;
      #pragma unroll
      for (int q = 0; q < 8; ++q)
        VtT[d8 + q][key] = Vt[cur][key][d8 + q];
    }

    // scores: 16 q-rows x 32 keys, contraction over dh=64 (2 k-chunks)
    v8f sc[2] = {};
    #pragma unroll
    for (int fn = 0; fn < 2; ++fn) {
      #pragma unroll
      for (int c = 0; c < 2; ++c) {
        v16bf bfr;
        int n = fn * 16 + l15;
        int kb = c * 32 + hl * 16;
        #pragma unroll
        for (int j = 0; j < 16; ++j) bfr[j] = Kt[cur][n][kb + j];
        sc[fn] = __builtin_amdgcn_wmma_f32_16x16x32_bf16(
            false, qf[c], false, bfr, (short)0, sc[fn], false, false);
      }
    }

    // online softmax (rows live across the 16-lane halves)
    float tmax[8];
    #pragma unroll
    for (int r = 0; r < 8; ++r) {
      float a = sc[0][r] * scale, c2 = sc[1][r] * scale;
      sc[0][r] = a; sc[1][r] = c2;
      tmax[r] = fmaxf(a, c2);
    }
    #pragma unroll
    for (int off = 1; off < 16; off <<= 1)
      #pragma unroll
      for (int r = 0; r < 8; ++r)
        tmax[r] = fmaxf(tmax[r], __shfl_xor(tmax[r], off, 32));

    float alpha[8], rsum[8];
    #pragma unroll
    for (int r = 0; r < 8; ++r) {
      float mn = fmaxf(mrow[r], tmax[r]);
      alpha[r] = __expf(mrow[r] - mn);
      mrow[r]  = mn;
      float p0 = __expf(sc[0][r] - mn);
      float p1 = __expf(sc[1][r] - mn);
      sc[0][r] = p0; sc[1][r] = p1;
      rsum[r] = p0 + p1;
    }
    #pragma unroll
    for (int off = 1; off < 16; off <<= 1)
      #pragma unroll
      for (int r = 0; r < 8; ++r)
        rsum[r] += __shfl_xor(rsum[r], off, 32);
    #pragma unroll
    for (int r = 0; r < 8; ++r) lrow[r] = lrow[r] * alpha[r] + rsum[r];
    #pragma unroll
    for (int nc = 0; nc < 4; ++nc)
      #pragma unroll
      for (int r = 0; r < 8; ++r) o[nc][r] *= alpha[r];

    // relayout P: D layout (lane=key, vgpr=row) -> A layout (wave-private LDS)
    #pragma unroll
    for (int fn = 0; fn < 2; ++fn)
      #pragma unroll
      for (int r = 0; r < 8; ++r)
        Ps[wave][r + 8 * hl][fn * 16 + l15] = f2bf(sc[fn][r]);
    v16bf pf;
    #pragma unroll
    for (int j = 0; j < 16; ++j) pf[j] = Ps[wave][l15][hl * 16 + j];

    __syncthreads();                     // VtT complete (and Kt reads done)

    // O(16x64) += P(16x32) @ V(32x64); fragments contiguous in VtT
    #pragma unroll
    for (int nc = 0; nc < 4; ++nc) {
      v16bf vf;
      int n = nc * 16 + l15;             // d index
      int kb = hl * 16;                  // key half
      #pragma unroll
      for (int j = 0; j < 16; ++j) vf[j] = VtT[n][kb + j];
      o[nc] = __builtin_amdgcn_wmma_f32_16x16x32_bf16(
          false, pf, false, vf, (short)0, o[nc], false, false);
    }
  }

  // epilogue: normalize and scatter (bf16)
  #pragma unroll
  for (int nc = 0; nc < 4; ++nc)
    #pragma unroll
    for (int r = 0; r < 8; ++r) {
      int m = q0 + r + 8 * hl;
      int n = nc * 16 + l15;
      out[((size_t)m * Bsz + b) * E + h * dh + n] = f2bf(o[nc][r] / lrow[r]);
    }
}

// ---------------------------------------------------------------------------
// Support kernels
// ---------------------------------------------------------------------------
__global__ __launch_bounds__(256)
void f32_to_bf16_kernel(const float* __restrict__ src, __bf16* __restrict__ dst,
                        size_t n) {
  size_t i = (size_t)blockIdx.x * 256 + threadIdx.x;
  if (i < n) dst[i] = f2bf(src[i]);
}

// cl_w (E x 30) -> bf16 padded to ld 32 with zeros
__global__ __launch_bounds__(256)
void convert_clw_kernel(const float* __restrict__ w, __bf16* __restrict__ d,
                        int E) {
  int i = blockIdx.x * 256 + threadIdx.x;     // E*32
  if (i >= E * 32) return;
  int r = i >> 5, c = i & 31;
  d[i] = (c < 30) ? f2bf(w[r * 30 + c]) : bf_zero();
}

__global__ __launch_bounds__(256)
void pe_add_kernel(float* __restrict__ x, int Bsz, int E) {
  size_t idx = (size_t)blockIdx.x * 256 + threadIdx.x;
  int e = (int)(idx % E);
  size_t n = idx / E;
  int s = (int)(n / Bsz);
  float de = (float)(2 * (e >> 1));
  float div = __expf(-de * (9.210340372f / (float)E));   // ln(10000)
  float a = (float)s * div;
  x[idx] += (e & 1) ? __cosf(a) : __sinf(a);
}

__global__ __launch_bounds__(256)
void trend_mean_kernel(const float* __restrict__ x, float* __restrict__ trendBE,
                       int S, int Bsz, int E) {
  int idx = blockIdx.x * 256 + threadIdx.x;     // b*E + e
  if (idx >= Bsz * E) return;
  float s = 0.f;
  for (int t = 0; t < S; ++t) s += x[(size_t)t * Bsz * E + idx];
  trendBE[idx] = s / (float)S;
}

// x -> seasonal (in place) + bf16 mirrors of seasonal & broadcast trend
__global__ __launch_bounds__(256)
void split_seasonal_kernel(float* __restrict__ x, const float* __restrict__ trendBE,
                           float* __restrict__ trendTok,
                           __bf16* __restrict__ xbf, __bf16* __restrict__ trendTokbf,
                           int Bsz, int E) {
  size_t idx = (size_t)blockIdx.x * 256 + threadIdx.x;
  int be = (int)(idx % ((size_t)Bsz * E));
  float t = trendBE[be];
  float xs = x[idx] - t;
  x[idx] = xs;
  xbf[idx] = f2bf(xs);
  trendTok[idx] = t;
  trendTokbf[idx] = f2bf(t);
}

// per-token mean over E -> sig[b*S + s]
__global__ __launch_bounds__(256)
void sig_mean_kernel(const float* __restrict__ y, float* __restrict__ sig,
                     int Bsz, int S, int E) {
  int token = (blockIdx.x * 256 + threadIdx.x) >> 5;
  int lane  = threadIdx.x & 31;
  if (token >= S * Bsz) return;
  const float* r = y + (size_t)token * E;
  float s = 0.f;
  for (int i = lane; i < E; i += 32) s += r[i];
  #pragma unroll
  for (int off = 1; off < 32; off <<= 1) s += __shfl_xor(s, off, 32);
  if (lane == 0) {
    int ss = token / Bsz, b = token % Bsz;
    sig[(size_t)b * S + ss] = s / (float)E;
  }
}

// Build int_psi and the 30 reversed CWT kernels on device (one block).
__global__ __launch_bounds__(256)
void cwt_setup_kernel(float* __restrict__ intpsi, float* __restrict__ kerbuf) {
  int tid = threadIdx.x;
  for (int i = tid; i < 1024; i += 256) {
    float xv = -8.f + 16.f * (float)i / 1023.f;
    intpsi[i] = __expf(-xv * xv * 0.5f) * __cosf(5.f * xv);
  }
  __syncthreads();
  if (tid == 0) {
    float step = 16.f / 1023.f, acc = 0.f;
    for (int i = 0; i < 1024; ++i) { acc += intpsi[i] * step; intpsi[i] = acc; }
  }
  __syncthreads();
  for (int sc = 1; sc <= 30; ++sc) {
    int len = 16 * sc + 1;
    float sstep = (float)sc * (16.f / 1023.f);
    for (int i = tid; i < len; i += 256) {
      int rev = len - 1 - i;                       // kernel is reversed
      int j = (int)((float)rev / sstep);
      if (j > 1023) j = 1023;
      kerbuf[(size_t)(sc - 1) * 512 + i] = intpsi[j];
    }
  }
}

// Direct conv + diff + center-crop -> bf16 dec0 (ld 32, cols 30/31 zeroed)
__global__ __launch_bounds__(512)
void cwt_conv_kernel(const float* __restrict__ sig, const float* __restrict__ kerbuf,
                     __bf16* __restrict__ dec0bf, int Bsz, int S) {
  int s  = threadIdx.x;           // 0..511
  int b  = blockIdx.x;
  int sc = blockIdx.y + 1;        // scale 1..30
  int len = 16 * sc + 1;
  int lo  = (len - 2) / 2;
  const float* k  = kerbuf + (size_t)(sc - 1) * 512;
  const float* sg = sig + (size_t)b * S;
  int t = s + lo;
  float c0 = 0.f, c1 = 0.f;
  for (int i = 0; i < len; ++i) {
    float kv = k[i];
    int u0 = t - i;
    int u1 = u0 + 1;
    if (u0 >= 0 && u0 < S) c0 += kv * sg[u0];
    if (u1 >= 0 && u1 < S) c1 += kv * sg[u1];
  }
  size_t rowb = ((size_t)s * Bsz + b) * 32;
  dec0bf[rowb + (sc - 1)] = f2bf(-sqrtf((float)sc) * (c1 - c0));
  if (sc == 1) { dec0bf[rowb + 30] = bf_zero(); dec0bf[rowb + 31] = bf_zero(); }
}

// residual + layer norm; warp per token (E=512 -> 16 vals/lane, wave32)
__global__ __launch_bounds__(256)
void ln_residual_kernel(const float* __restrict__ x, const float* __restrict__ res,
                        const float* __restrict__ g, const float* __restrict__ bb,
                        float* __restrict__ dst, __bf16* __restrict__ dstbf,
                        int Ntok, int E) {
  int token = (blockIdx.x * 256 + threadIdx.x) >> 5;
  int lane  = threadIdx.x & 31;
  if (token >= Ntok) return;
  const float* xr = x   + (size_t)token * E;
  const float* rr = res + (size_t)token * E;
  float v[16];
  float s = 0.f;
  #pragma unroll
  for (int i = 0; i < 16; ++i) { v[i] = xr[lane + i * 32] + rr[lane + i * 32]; s += v[i]; }
  #pragma unroll
  for (int off = 1; off < 32; off <<= 1) s += __shfl_xor(s, off, 32);
  float mu = s / (float)E;
  float var = 0.f;
  #pragma unroll
  for (int i = 0; i < 16; ++i) { float d = v[i] - mu; var += d * d; }
  #pragma unroll
  for (int off = 1; off < 32; off <<= 1) var += __shfl_xor(var, off, 32);
  float inv = rsqrtf(var / (float)E + 1e-5f);
  #pragma unroll
  for (int i = 0; i < 16; ++i) {
    int col = lane + i * 32;
    float o = (v[i] - mu) * inv * g[col] + bb[col];
    dst[(size_t)token * E + col] = o;
    if (dstbf) dstbf[(size_t)token * E + col] = f2bf(o);
  }
}

// ---------------------------------------------------------------------------
// Host orchestration
// ---------------------------------------------------------------------------
extern "C" void kernel_launch(void* const* d_in, const int* in_sizes, int n_in,
                              void* d_out, int out_size, void* d_ws, size_t ws_size,
                              hipStream_t stream) {
  (void)in_sizes; (void)n_in; (void)out_size; (void)ws_size;
  const int S = 512, Bsz = 16, D = 64, E = 512, FF = 2048, L = 4, H = 8;
  const int N = S * Bsz;          // 8192 tokens
  const int E3 = 3 * E;           // 1536

  const float* src     = (const float*)d_in[0];
  const float* emb_w   = (const float*)d_in[1];
  const float* emb_b   = (const float*)d_in[2];
  const float* t_in_w  = (const float*)d_in[3];
  const float* t_in_b  = (const float*)d_in[4];
  const float* t_out_w = (const float*)d_in[5];
  const float* t_out_b = (const float*)d_in[6];
  const float* c_in_w  = (const float*)d_in[7];
  const float* c_in_b  = (const float*)d_in[8];
  const float* c_out_w = (const float*)d_in[9];
  const float* c_out_b = (const float*)d_in[10];
  const float* cl_w    = (const float*)d_in[11];
  const float* cl_b    = (const float*)d_in[12];
  const float* enc_in_w  = (const float*)d_in[13];
  const float* enc_in_b  = (const float*)d_in[14];
  const float* enc_out_w = (const float*)d_in[15];
  const float* enc_out_b = (const float*)d_in[16];
  const float* ff1_w   = (const float*)d_in[17];
  const float* ff1_b   = (const float*)d_in[18];
  const float* ff2_w   = (const float*)d_in[19];
  const float* ff2_b   = (const float*)d_in[20];
  const float* ln1_g   = (const float*)d_in[21];
  const float* ln1_b   = (const float*)d_in[22];
  const float* ln2_g   = (const float*)d_in[23];
  const float* ln2_b   = (const float*)d_in[24];
  float* out_final = (float*)d_out;

  // ---- workspace carve-up (deterministic) ----
  char* w = (char*)d_ws;
  auto carve = [&](size_t bytes) -> char* {
    char* p = w; w += (bytes + 255) & ~(size_t)255; return p;
  };
  // fp32 activations
  float*  xbuf     = (float*) carve((size_t)N * E * 4);      // x -> seasonal
  float*  trendBE  = (float*) carve((size_t)Bsz * E * 4);
  float*  trendTok = (float*) carve((size_t)N * E * 4);
  float*  combined = (float*) carve((size_t)N * E * 4);      // encoder state
  float*  tmp      = (float*) carve((size_t)N * E * 4);
  float*  sig      = (float*) carve((size_t)Bsz * S * 4);
  float*  intpsi   = (float*) carve(1024 * 4);
  float*  kerbuf   = (float*) carve((size_t)30 * 512 * 4);
  // bf16 activations
  __bf16* srcbf    = (__bf16*)carve((size_t)N * D * 2);
  __bf16* xbf      = (__bf16*)carve((size_t)N * E * 2);
  __bf16* trendTbf = (__bf16*)carve((size_t)N * E * 2);
  __bf16* outbf    = (__bf16*)carve((size_t)N * E * 2);
  __bf16* qkvbf    = (__bf16*)carve((size_t)N * E3 * 2);
  __bf16* attnbf   = (__bf16*)carve((size_t)N * E * 2);
  __bf16* ffbf     = (__bf16*)carve((size_t)N * FF * 2);
  __bf16* decEbf   = (__bf16*)carve((size_t)N * E * 2);
  __bf16* dec0bf   = (__bf16*)carve((size_t)N * 32 * 2);
  // bf16 weights
  __bf16* emb_wbf   = (__bf16*)carve((size_t)E * D * 2);
  __bf16* t_in_wbf  = (__bf16*)carve((size_t)E3 * E * 2);
  __bf16* t_out_wbf = (__bf16*)carve((size_t)E * E * 2);
  __bf16* c_in_wbf  = (__bf16*)carve((size_t)E3 * E * 2);
  __bf16* c_out_wbf = (__bf16*)carve((size_t)E * E * 2);
  __bf16* cl_wbf    = (__bf16*)carve((size_t)E * 32 * 2);
  __bf16* enc_in_wbf  = (__bf16*)carve((size_t)L * E3 * E * 2);
  __bf16* enc_out_wbf = (__bf16*)carve((size_t)L * E * E * 2);
  __bf16* ff1_wbf   = (__bf16*)carve((size_t)L * FF * E * 2);
  __bf16* ff2_wbf   = (__bf16*)carve((size_t)L * E * FF * 2);

  auto tobf = [&](const float* s, __bf16* d, size_t n) {
    f32_to_bf16_kernel<<<(unsigned)((n + 255) / 256), 256, 0, stream>>>(s, d, n);
  };
  auto gemm = [&](const __bf16* A, const __bf16* Bm, const float* bias,
                  float* C, __bf16* Cbf,
                  int M, int Nn, int K, int lda, int ldb, int ldc, int flags) {
    dim3 grid(M / GBM, Nn / GBN);
    gemm_bf16_wmma<<<grid, 256, 0, stream>>>(A, Bm, bias, C, Cbf,
                                             M, Nn, K, lda, ldb, ldc, flags);
  };
  auto flash = [&]() {
    attn_flash_wmma<<<dim3(S / 128, Bsz, H), 256, 0, stream>>>(qkvbf, attnbf,
                                                               Bsz, S, E, H);
  };

  // ---- one-time bf16 conversions ----
  tobf(src, srcbf, (size_t)N * D);
  tobf(emb_w, emb_wbf, (size_t)E * D);
  tobf(t_in_w,  t_in_wbf,  (size_t)E3 * E);
  tobf(t_out_w, t_out_wbf, (size_t)E * E);
  tobf(c_in_w,  c_in_wbf,  (size_t)E3 * E);
  tobf(c_out_w, c_out_wbf, (size_t)E * E);
  convert_clw_kernel<<<(E * 32 + 255) / 256, 256, 0, stream>>>(cl_w, cl_wbf, E);
  tobf(enc_in_w,  enc_in_wbf,  (size_t)L * E3 * E);
  tobf(enc_out_w, enc_out_wbf, (size_t)L * E * E);
  tobf(ff1_w, ff1_wbf, (size_t)L * FF * E);
  tobf(ff2_w, ff2_wbf, (size_t)L * E * FF);

  // ---- embedding + positional encoding ----
  gemm(srcbf, emb_wbf, emb_b, xbuf, nullptr, N, E, D, D, D, E, 0);
  pe_add_kernel<<<(unsigned)((size_t)N * E / 256), 256, 0, stream>>>(xbuf, Bsz, E);

  // ---- trend / seasonal split (+ bf16 mirrors) ----
  trend_mean_kernel<<<(Bsz * E + 255) / 256, 256, 0, stream>>>(xbuf, trendBE, S, Bsz, E);
  split_seasonal_kernel<<<(unsigned)((size_t)N * E / 256), 256, 0, stream>>>(
      xbuf, trendBE, trendTok, xbf, trendTbf, Bsz, E);

  // ---- CWT tables ----
  cwt_setup_kernel<<<1, 256, 0, stream>>>(intpsi, kerbuf);

  auto inpar = [&](const float* y, const __bf16* ybf, bool first) {
    // t-branch MHA
    gemm(ybf, t_in_wbf, t_in_b, nullptr, qkvbf, N, E3, E, E, E, E3, 0);
    flash();
    gemm(attnbf, t_out_wbf, t_out_b, combined, nullptr, N, E, E, E, E, E,
         first ? 0 : 1);
    // CWT branch
    sig_mean_kernel<<<(N * 32 + 255) / 256, 256, 0, stream>>>(y, sig, Bsz, S, E);
    cwt_conv_kernel<<<dim3(Bsz, 30), 512, 0, stream>>>(sig, kerbuf, dec0bf, Bsz, S);
    gemm(dec0bf, cl_wbf, cl_b, nullptr, decEbf, N, E, 32, 32, 32, E, 0);
    gemm(decEbf, c_in_wbf, c_in_b, nullptr, qkvbf, N, E3, E, E, E, E3, 0);
    flash();
    gemm(attnbf, c_out_wbf, c_out_b, combined, nullptr, N, E, E, E, E, E, 1);
  };
  inpar(xbuf, xbf, true);           // seasonal
  inpar(trendTok, trendTbf, false); // trend

  // encoder state: fp32 `combined` + bf16 mirror
  tobf(combined, outbf, (size_t)N * E);

  // ---- encoder stack ----
  float* out = combined;
  for (int l = 0; l < L; ++l) {
    const __bf16* wq = enc_in_wbf  + (size_t)l * E3 * E;
    const float*  bq = enc_in_b    + (size_t)l * E3;
    const __bf16* wo = enc_out_wbf + (size_t)l * E * E;
    const float*  bo = enc_out_b   + (size_t)l * E;
    const __bf16* w1 = ff1_wbf + (size_t)l * FF * E;
    const float*  b1 = ff1_b   + (size_t)l * FF;
    const __bf16* w2 = ff2_wbf + (size_t)l * E * FF;
    const float*  b2 = ff2_b   + (size_t)l * E;

    gemm(outbf, wq, bq, nullptr, qkvbf, N, E3, E, E, E, E3, 0);
    flash();
    gemm(attnbf, wo, bo, tmp, nullptr, N, E, E, E, E, E, 0);
    ln_residual_kernel<<<(N * 32 + 255) / 256, 256, 0, stream>>>(
        out, tmp, ln1_g + (size_t)l * E, ln1_b + (size_t)l * E, out, outbf, N, E);

    gemm(outbf, w1, b1, nullptr, ffbf, N, FF, E, E, E, FF, 2 /*relu*/);
    gemm(ffbf, w2, b2, tmp, nullptr, N, E, FF, FF, FF, E, 0);
    float*  dst   = (l == L - 1) ? out_final : out;
    __bf16* dstbf = (l == L - 1) ? nullptr   : outbf;
    ln_residual_kernel<<<(N * 32 + 255) / 256, 256, 0, stream>>>(
        out, tmp, ln2_g + (size_t)l * E, ln2_b + (size_t)l * E, dst, dstbf, N, E);
  }
}
